// DAML_75007308857754
// MI455X (gfx1250) — compile-verified, matrix operand order checked by reading, not run on hardware
//
#include <hip/hip_runtime.h>

typedef unsigned int u32;
typedef unsigned short u16;
typedef __attribute__((ext_vector_type(16))) __bf16 v16bf;
typedef __attribute__((ext_vector_type(8)))  float  v8f;
typedef __attribute__((ext_vector_type(8)))  u32    v8u;

// ---------------- constants ----------------
#define BB   128   // batch
#define LL   512   // doc length
#define DD   128   // word dim
#define FF   100   // filters
#define FP   112   // filters padded to 7*16
#define KIC  384   // 3*D im2col contraction
#define LP   514   // L + 2 halo rows
#define IDD  32

// ---------------- helpers ----------------
__device__ __forceinline__ u16 f2bf(float x) {
  u32 u = __builtin_bit_cast(u32, x);
  u32 r = u + 0x7FFFu + ((u >> 16) & 1u);   // round-to-nearest-even
  return (u16)(r >> 16);
}
__device__ __forceinline__ u32 pack2(float a, float b) {
  return (u32)f2bf(a) | ((u32)f2bf(b) << 16);
}
// 16-bit A/B fragment K map (uint index within a 16-uint k-block):
//   k = (v<4 ? 2v : 16+2(v-4)) + 8*half  ->  k/2 = (v<4 ? v : 8+(v-4)) + 4*half
__device__ __forceinline__ int KH(int v, int half) {
  return ((v < 4) ? v : 8 + (v - 4)) + 4 * half;
}
__device__ __forceinline__ v8f wmma_bf16(v8u a, v8u b, v8f c) {
  v16bf av = __builtin_bit_cast(v16bf, a);
  v16bf bv = __builtin_bit_cast(v16bf, b);
  return __builtin_amdgcn_wmma_f32_16x16x32_bf16(false, av, false, bv,
                                                 (short)0, c, false, false);
}
__device__ __forceinline__ float wave_sum32(float v) {
#pragma unroll
  for (int m = 16; m >= 1; m >>= 1) v += __shfl_xor(v, m, 32);
  return v;
}
// CDNA5 async global->LDS copy (16B per lane), tracked by ASYNCcnt.
// dst = LDS byte offset (low 32 bits of a generic LDS pointer), src = global.
__device__ __forceinline__ void async_g2l_b128(u32 lds_off, const void* src) {
  asm volatile("global_load_async_to_lds_b128 %0, %1, off"
               :: "v"(lds_off), "v"(src) : "memory");
}
__device__ __forceinline__ void async_wait0() {
  asm volatile("s_wait_asynccnt 0" ::: "memory");
}
__device__ __forceinline__ u32 lds_off32(const void* p) {
  return (u32)(uintptr_t)p;   // LDS aperture: addr[31:0] is the LDS byte offset
}

extern __shared__ char dyn_lds[];

// ---------------- kernel 0: weight pack + pad zeroing ----------------
__global__ void prep_kernel(const float* __restrict__ dcw_u,
                            const float* __restrict__ dcw_i,
                            u16* __restrict__ Wbf,
                            u32* __restrict__ ebuf_u, u32* __restrict__ ebuf_i,
                            u32* __restrict__ pk_u,   u32* __restrict__ pk_i) {
  int tid = blockIdx.x * blockDim.x + threadIdx.x;
  int stride = gridDim.x * blockDim.x;
  // pack doc_cnn weights to bf16, zero-pad F=100 -> 112
  for (int i = tid; i < 2 * FP * KIC; i += stride) {
    int side = i / (FP * KIC);
    int rem  = i % (FP * KIC);
    int f = rem / KIC, kk = rem % KIC;
    const float* w = side ? dcw_i : dcw_u;
    Wbf[i] = (f < FF) ? f2bf(w[f * KIC + kk]) : (u16)0;
  }
  // zero halo rows (0 and 513) of the gated-embedding buffers
  for (int i = tid; i < 2 * BB * 2 * 64; i += stride) {
    int side = i / (BB * 2 * 64);
    int rem  = i % (BB * 2 * 64);
    int b = rem / (2 * 64), r2 = rem % (2 * 64);
    int row = (r2 / 64) ? (LP - 1) : 0;
    int j = r2 % 64;
    u32* e = side ? ebuf_i : ebuf_u;
    e[((size_t)b * LP + row) * 64 + j] = 0u;
  }
  // zero packed-fea K rows 56..63 (f = 112..127 pad of the K=128 cross GEMM)
  for (int i = tid; i < 2 * BB * 8 * LL; i += stride) {
    int side = i / (BB * 8 * LL);
    int rem  = i % (BB * 8 * LL);
    int b = rem / (8 * LL);
    int r = (rem / LL) % 8;
    int l = rem % LL;
    u32* p = side ? pk_i : pk_u;
    p[((size_t)b * 64 + 56 + r) * LL + l] = 0u;
  }
}

// ---------------- kernel 1: gather + word_cnn gate -> bf16 gated emb ----------------
__global__ __launch_bounds__(256) void embed_gate_kernel(
    const int* __restrict__ doc, const float* __restrict__ emb,
    const float* __restrict__ wcw, const float* __restrict__ wcb,
    u32* __restrict__ ebuf) {
  __shared__ float s_emb[68 * 128];   // rows l0-2 .. l0+65
  __shared__ float s_w[640];
  int b = blockIdx.x, l0 = blockIdx.y * 64, t = threadIdx.x;
#pragma unroll 2
  for (int i = 0; i < 34; ++i) {      // 68*128 = 8704 = 34*256
    int idx = t + i * 256;
    int row = idx >> 7, d = idx & 127;
    int gl = l0 - 2 + row;
    float v = 0.f;
    if (gl >= 0 && gl < LL) v = emb[(size_t)doc[b * LL + gl] * DD + d];
    s_emb[idx] = v;
  }
  for (int i = t; i < 640; i += 256) s_w[i] = wcw[i];
  __syncthreads();
  int w = t >> 5, lane = t & 31;
  float bias = wcb[0];
  for (int j = 0; j < 8; ++j) {
    int r = w * 8 + j;
    int l = l0 + r;
    int off = r * 128;                 // window rows l-2..l+2 are LDS rows r..r+4
    float acc = 0.f;
#pragma unroll
    for (int it = 0; it < 20; ++it) {
      int tt = lane + it * 32;
      acc += s_emb[off + tt] * s_w[tt];
    }
    acc = wave_sum32(acc);
    float g = 1.f / (1.f + __expf(-(acc + bias)));
    int d0 = lane * 4;
    const float* er = &s_emb[off + 256];          // row l itself
    u32 p0 = pack2(er[d0] * g, er[d0 + 1] * g);
    u32 p1 = pack2(er[d0 + 2] * g, er[d0 + 3] * g);
    size_t ob = ((size_t)b * LP + l + 1) * 64 + lane * 2;
    ebuf[ob] = p0; ebuf[ob + 1] = p1;
  }
}

// ---------------- kernel 2: doc_cnn as WMMA GEMM (M=F, N=l, K=384) ----------------
// LDS staged with CDNA5 async global->LDS B128 copies (ASYNCcnt).
__global__ __launch_bounds__(256) void doc_cnn_kernel(
    const u32* __restrict__ WbfU, const float* __restrict__ dcb,
    const u32* __restrict__ ebuf, float* __restrict__ fea,
    u32* __restrict__ pk) {
  u32* s_w = (u32*)dyn_lds;           // 112*192 = 21504 uints (bf16 weights)
  u32* s_e = s_w + 21504;             // 130*64  =  8320 uints (gated emb slab)
  int b = blockIdx.x, Lb = blockIdx.y * 128, t = threadIdx.x;
  {
    u32 swoff = lds_off32(s_w);
    const char* wsrc = (const char*)WbfU;
    for (int i = t; i < 5376; i += 256)           // 86016 B in 16B chunks
      async_g2l_b128(swoff + (u32)i * 16u, wsrc + (size_t)i * 16);
    u32 seoff = lds_off32(s_e);
    const char* esrc = (const char*)(ebuf + ((size_t)b * LP + Lb) * 64);
    for (int i = t; i < 2080; i += 256)           // 33280 B in 16B chunks
      async_g2l_b128(seoff + (u32)i * 16u, esrc + (size_t)i * 16);
    async_wait0();
  }
  __syncthreads();
  int w = t >> 5, lane = t & 31, n = lane & 15, half = lane >> 4;
  int l = Lb + w * 16 + n;
  v8f c[7];
#pragma unroll
  for (int ft = 0; ft < 7; ++ft)
#pragma unroll
    for (int r = 0; r < 8; ++r) {
      int f = ft * 16 + r + 8 * half;
      c[ft][r] = (f < FF) ? dcb[f] : 0.f;
    }
  int ebase = (w * 16 + n) * 64;      // im2col row l = contiguous 384 bf16
  for (int ks = 0; ks < 12; ++ks) {
    v8u bf;
#pragma unroll
    for (int v = 0; v < 8; ++v) bf[v] = s_e[ebase + ks * 16 + KH(v, half)];
#pragma unroll
    for (int ft = 0; ft < 7; ++ft) {
      int frow = ft * 16 + n;
      v8u af;
#pragma unroll
      for (int v = 0; v < 8; ++v) af[v] = s_w[frow * 192 + ks * 16 + KH(v, half)];
      c[ft] = wmma_bf16(af, bf, c[ft]);
    }
  }
#pragma unroll
  for (int ft = 0; ft < 7; ++ft) {
#pragma unroll
    for (int r = 0; r < 8; ++r) {
      int f = ft * 16 + r + 8 * half;
      fea[((size_t)b * FP + f) * LL + l] = c[ft][r];
    }
#pragma unroll
    for (int j = 0; j < 4; ++j) {     // K-pair-packed B layout for cross GEMM
      int fp = ft * 8 + j + 4 * half; // = (ft*16 + 2j + 8*half)/2
      pk[((size_t)b * 64 + fp) * LL + l] = pack2(c[ft][2 * j], c[ft][2 * j + 1]);
    }
  }
}

// ---------------- kernel 3: fea (b,f,l) fp32 -> A-layout bf16 (b,l,128) ----------------
__global__ __launch_bounds__(256) void transpose_kernel(
    const float* __restrict__ fea, u32* __restrict__ feaT) {
  __shared__ float tile[128][104];
  int b = blockIdx.x, Lb = blockIdx.y * 128, t = threadIdx.x;
#pragma unroll 2
  for (int i = 0; i < 50; ++i) {      // 100*128 = 12800 = 50*256
    int idx = t + i * 256;
    int f = idx / 128, l = idx & 127;
    tile[l][f] = fea[((size_t)b * FP + f) * LL + Lb + l];
  }
  __syncthreads();
#pragma unroll 2
  for (int i = 0; i < 32; ++i) {      // 128*64 = 8192 = 32*256
    int idx = t + i * 256;
    int l = idx >> 6, j = idx & 63;
    int f = 2 * j;
    float a  = (f < FF)     ? tile[l][f]     : 0.f;
    float bb = (f + 1 < FF) ? tile[l][f + 1] : 0.f;
    feaT[((size_t)b * LL + Lb + l) * 64 + j] = pack2(a, bb);
  }
}

// ---------------- kernel 4: squared filter norms ----------------
__global__ void norm_kernel(const float* __restrict__ fea, float* __restrict__ n2) {
  int b = blockIdx.x;
  int l = blockIdx.y * 256 + threadIdx.x;
  float s = 0.f;
  for (int f = 0; f < FF; ++f) {
    float v = fea[((size_t)b * FP + f) * LL + l];
    s += v * v;
  }
  n2[(size_t)b * LL + l] = s;
}

// ---------------- kernel 5: fused cross GEMM + att + row-sum ----------------
// Full 64x512 packed-B panel staged once per block via async B128 -> LDS.
__global__ __launch_bounds__(256) void cross_att_kernel(
    const u32* __restrict__ featA, const u32* __restrict__ pkB,
    const float* __restrict__ a2, const float* __restrict__ b2,
    float* __restrict__ attOut) {
  u32* s_pb = (u32*)dyn_lds;          // 64*512 uints = 128 KB
  int b = blockIdx.x, Lb = blockIdx.y * 128, t = threadIdx.x;
  {
    u32 pboff = lds_off32(s_pb);
    const char* src = (const char*)(pkB + (size_t)b * 64 * LL);
    for (int i = t; i < 8192; i += 256)           // 131072 B in 16B chunks
      async_g2l_b128(pboff + (u32)i * 16u, src + (size_t)i * 16);
  }
  int lane = t & 31, n = lane & 15, half = lane >> 4;
  int lb = Lb + (t >> 5) * 16;
  v8u aU[4];
  const u32* fA = featA + ((size_t)b * LL + lb + n) * 64;
#pragma unroll
  for (int kb = 0; kb < 4; ++kb)
#pragma unroll
    for (int v = 0; v < 8; ++v) aU[kb][v] = fA[kb * 16 + KH(v, half)];
  float u2v[8], rowacc[8];
#pragma unroll
  for (int r = 0; r < 8; ++r) {
    u2v[r] = a2[(size_t)b * LL + lb + r + 8 * half];
    rowacc[r] = 0.f;
  }
  async_wait0();
  __syncthreads();
  for (int mt = 0; mt < 32; ++mt) {
    int m = mt * 16 + n;
    float i2m = b2[(size_t)b * LL + m];
    v8f c;
#pragma unroll
    for (int r = 0; r < 8; ++r) c[r] = 0.f;
#pragma unroll
    for (int kb = 0; kb < 4; ++kb) {
      v8u bfv;
#pragma unroll
      for (int v = 0; v < 8; ++v)
        bfv[v] = s_pb[(kb * 16 + KH(v, half)) * LL + m];
      c = wmma_bf16(aU[kb], bfv, c);
    }
#pragma unroll
    for (int r = 0; r < 8; ++r) {
      float sq = fmaxf(u2v[r] + i2m - 2.f * c[r], 1e-12f);
      rowacc[r] += 1.f / (1.f + sqrtf(sq));
    }
  }
#pragma unroll
  for (int r = 0; r < 8; ++r) {
    float v = rowacc[r];
#pragma unroll
    for (int mk = 8; mk >= 1; mk >>= 1) v += __shfl_xor(v, mk, 32);
    if (n == 0) attOut[(size_t)b * LL + lb + r + 8 * half] = v;
  }
}

// ---------------- kernel 6: collapsed pool+abs_cnn window sums S_k ----------------
__global__ void pool_reduce_kernel(const float* __restrict__ fea,
                                   const float* __restrict__ att,
                                   float* __restrict__ S) {
  int b = blockIdx.x, t = threadIdx.x, w = t >> 5, lane = t & 31;
  const float* ab = att + (size_t)b * LL;
  for (int g = w; g < FF; g += 8) {
    const float* fb = fea + ((size_t)b * FP + g) * LL;
    if (g + 8 < FF)                              // gfx1250 global_prefetch_b8
      __builtin_prefetch(fb + 8 * LL + lane * 16, 0, 1);
    float s = 0.f;
#pragma unroll 4
    for (int it = 0; it < 16; ++it) {
      int l = lane + it * 32;
      s += fb[l] * ab[l];
    }
    s = wave_sum32(s);
    if (lane == 0) {
      float w0 = fb[0] * ab[0],   w1 = fb[1] * ab[1],   w2 = fb[2] * ab[2];
      float e3 = fb[509] * ab[509], e2 = fb[510] * ab[510], e1 = fb[511] * ab[511];
      float T = 3.f * s - w0 - e1;          // sum_t pools[t]
      float p0 = w0 + w1, p1 = p0 + w2;
      float q2 = e3 + e2 + e1, q1 = e2 + e1;
      S[((size_t)b * 3 + 0) * FP + g] = T - q2 - q1;
      S[((size_t)b * 3 + 1) * FP + g] = T - p0 - q1;
      S[((size_t)b * 3 + 2) * FP + g] = T - p0 - p1;
    }
  }
}

// ---------------- kernel 7: abs_cnn-mean + FC + prediction ----------------
__global__ __launch_bounds__(128) void final_kernel(
    const float* __restrict__ S_u, const float* __restrict__ S_i,
    const float* __restrict__ absw_u, const float* __restrict__ absb_u,
    const float* __restrict__ absw_i, const float* __restrict__ absb_i,
    const float* __restrict__ fcw_u, const float* __restrict__ fcb_u,
    const float* __restrict__ fcw_i, const float* __restrict__ fcb_i,
    const float* __restrict__ uid_emb, const float* __restrict__ iid_emb,
    const float* __restrict__ predw, const float* __restrict__ predb,
    const float* __restrict__ ubias, const float* __restrict__ ibias,
    const float* __restrict__ gbias,
    const int* __restrict__ uids, const int* __restrict__ iids,
    float* __restrict__ out) {
  __shared__ float ybu[FF], ybi[FF], ud[IDD], idv[IDD], red[128];
  int b = blockIdx.x, t = threadIdx.x;
  if (t < FF) {
    float au = 0.f, ai = 0.f;
    for (int k = 0; k < 3; ++k)
      for (int g = 0; g < FF; ++g) {
        float su = S_u[((size_t)b * 3 + k) * FP + g];
        float si = S_i[((size_t)b * 3 + k) * FP + g];
        au += absw_u[t * 300 + k * 100 + g] * su;
        ai += absw_i[t * 300 + k * 100 + g] * si;
      }
    ybu[t] = au * (1.f / 510.f) + absb_u[t];
    ybi[t] = ai * (1.f / 510.f) + absb_i[t];
  }
  __syncthreads();
  if (t < IDD) {
    float su = fcb_u[t], si = fcb_i[t];
    for (int f = 0; f < FF; ++f) {
      su += fcw_u[t * FF + f] * ybu[f];
      si += fcw_i[t * FF + f] * ybi[f];
    }
    ud[t]  = fmaxf(su, 0.f);
    idv[t] = fmaxf(si, 0.f);
  }
  __syncthreads();
  int uid = uids[b], iid = iids[b];
  float j;
  if      (t < 32) j = ud[t];
  else if (t < 64) j = uid_emb[(size_t)uid * IDD + (t - 32)];
  else if (t < 96) j = idv[t - 64];
  else             j = iid_emb[(size_t)iid * IDD + (t - 96)];
  red[t] = predw[t] * j;
  __syncthreads();
  if (t == 0) {
    float s = 0.f;
    for (int i = 0; i < 128; ++i) s += red[i];
    out[b] = s + predb[0] + ubias[uid] + ibias[iid] + gbias[0];
  }
}

// ---------------- launcher ----------------
extern "C" void kernel_launch(void* const* d_in, const int* in_sizes, int n_in,
                              void* d_out, int out_size, void* d_ws, size_t ws_size,
                              hipStream_t stream) {
  (void)in_sizes; (void)n_in; (void)out_size; (void)ws_size;
  const int*   user_doc   = (const int*)d_in[0];
  const int*   item_doc   = (const int*)d_in[1];
  const int*   uids       = (const int*)d_in[2];
  const int*   iids       = (const int*)d_in[3];
  const float* u_word_emb = (const float*)d_in[4];
  const float* i_word_emb = (const float*)d_in[5];
  const float* wcw        = (const float*)d_in[6];
  const float* wcb        = (const float*)d_in[7];
  const float* u_doc_w    = (const float*)d_in[8];
  const float* u_doc_b    = (const float*)d_in[9];
  const float* i_doc_w    = (const float*)d_in[10];
  const float* i_doc_b    = (const float*)d_in[11];
  const float* u_abs_w    = (const float*)d_in[12];
  const float* u_abs_b    = (const float*)d_in[13];
  const float* i_abs_w    = (const float*)d_in[14];
  const float* i_abs_b    = (const float*)d_in[15];
  const float* u_fc_w     = (const float*)d_in[16];
  const float* u_fc_b     = (const float*)d_in[17];
  const float* i_fc_w     = (const float*)d_in[18];
  const float* i_fc_b     = (const float*)d_in[19];
  const float* uid_emb    = (const float*)d_in[20];
  const float* iid_emb    = (const float*)d_in[21];
  const float* pred_w     = (const float*)d_in[22];
  const float* pred_b     = (const float*)d_in[23];
  const float* user_bias  = (const float*)d_in[24];
  const float* item_bias  = (const float*)d_in[25];
  const float* glob_bias  = (const float*)d_in[26];
  float* out = (float*)d_out;

  char* ws = (char*)d_ws;
  size_t off = 0;
  auto take = [&](size_t bytes) -> char* {
    char* p = ws + off;
    off += (bytes + 255) & ~(size_t)255;
    return p;
  };
  constexpr size_t SZ_EBUF = (size_t)BB * LP * DD * 2;   // bf16
  constexpr size_t SZ_WBF  = (size_t)2 * FP * KIC * 2;   // bf16
  constexpr size_t SZ_FEA  = (size_t)BB * FP * LL * 4;   // f32
  constexpr size_t SZ_PK   = (size_t)BB * 64 * LL * 4;   // u32 (bf16 pairs)
  constexpr size_t SZ_FT   = (size_t)BB * LL * 64 * 4;   // u32 (bf16 pairs)
  constexpr size_t SZ_N2   = (size_t)BB * LL * 4;
  constexpr size_t SZ_S    = (size_t)BB * 3 * FP * 4;

  u32* ebuf_u = (u32*)take(SZ_EBUF);
  u32* ebuf_i = (u32*)take(SZ_EBUF);
  u16* Wbf    = (u16*)take(SZ_WBF);
  float* fea_u = (float*)take(SZ_FEA);
  float* fea_i = (float*)take(SZ_FEA);
  u32* pk_u   = (u32*)take(SZ_PK);
  u32* pk_i   = (u32*)take(SZ_PK);
  u32* feaT_u = (u32*)take(SZ_FT);
  u32* feaT_i = (u32*)take(SZ_FT);
  float* u2   = (float*)take(SZ_N2);
  float* i2   = (float*)take(SZ_N2);
  float* uatt = (float*)take(SZ_N2);
  float* iatt = (float*)take(SZ_N2);
  float* S_u  = (float*)take(SZ_S);
  float* S_i  = (float*)take(SZ_S);

  const u32* WbfU = (const u32*)Wbf;
  constexpr unsigned DOC_LDS   = (21504 + 8320) * 4;   // 119,296 B
  constexpr unsigned CROSS_LDS = 64 * 512 * 4;         // 131,072 B

  prep_kernel<<<dim3(512), 256, 0, stream>>>(u_doc_w, i_doc_w, Wbf,
                                             ebuf_u, ebuf_i, pk_u, pk_i);

  embed_gate_kernel<<<dim3(BB, LL / 64), 256, 0, stream>>>(user_doc, u_word_emb, wcw, wcb, ebuf_u);
  embed_gate_kernel<<<dim3(BB, LL / 64), 256, 0, stream>>>(item_doc, i_word_emb, wcw, wcb, ebuf_i);

  doc_cnn_kernel<<<dim3(BB, LL / 128), 256, DOC_LDS, stream>>>(WbfU,         u_doc_b, ebuf_u, fea_u, pk_u);
  doc_cnn_kernel<<<dim3(BB, LL / 128), 256, DOC_LDS, stream>>>(WbfU + 21504, i_doc_b, ebuf_i, fea_i, pk_i);

  transpose_kernel<<<dim3(BB, LL / 128), 256, 0, stream>>>(fea_u, feaT_u);
  transpose_kernel<<<dim3(BB, LL / 128), 256, 0, stream>>>(fea_i, feaT_i);

  norm_kernel<<<dim3(BB, LL / 256), 256, 0, stream>>>(fea_u, u2);
  norm_kernel<<<dim3(BB, LL / 256), 256, 0, stream>>>(fea_i, i2);

  // user_att = row sums of att(u,i); item_att = row sums with roles swapped
  cross_att_kernel<<<dim3(BB, LL / 128), 256, CROSS_LDS, stream>>>(feaT_u, pk_i, u2, i2, uatt);
  cross_att_kernel<<<dim3(BB, LL / 128), 256, CROSS_LDS, stream>>>(feaT_i, pk_u, i2, u2, iatt);

  pool_reduce_kernel<<<dim3(BB), 256, 0, stream>>>(fea_u, uatt, S_u);
  pool_reduce_kernel<<<dim3(BB), 256, 0, stream>>>(fea_i, iatt, S_i);

  final_kernel<<<dim3(BB), 128, 0, stream>>>(S_u, S_i, u_abs_w, u_abs_b, i_abs_w, i_abs_b,
                                             u_fc_w, u_fc_b, i_fc_w, i_fc_b,
                                             uid_emb, iid_emb, pred_w, pred_b,
                                             user_bias, item_bias, glob_bias,
                                             uids, iids, out);
}
